// LSTMAutoencoder_52956946759974
// MI455X (gfx1250) — matrix operand order repaired and checked by reading, not compiled, not run on hardware
//
#include <hip/hip_runtime.h>

// LSTM autoencoder, B=128 T=512 I=64 H=512 L=2 — bf16 WMMA, f32 accumulate.
// 8 workgroups (one per 16-batch slice, recurrence stays WGP-local: only
// s_barrier between phases). 16 waves/WG; each wave owns 32 h columns =
// 8 gate M-tiles {i,f,g,o} x 2. gates^T = W x h^T: B-operand = activations
// (ds_load_b128 from LDS / global_load_b128 from HBM, 32B/lane contiguous),
// A-operand = pre-packed weight fragments streamed from L2 with 4-tile
// ping-pong software pipelining (round-2 structure: no scratch spills,
// staggered s_wait_loadcnt).

#define B_    128
#define T_    512
#define I_    64
#define H_    512
#define G4_   2048
#define NTHR  512
#define HSTR  1056   // LDS h row stride (bytes): 512*2 + 32 pad
#define YSTR  160    // LDS y row stride (bytes): 64*2 + 32 pad

typedef __bf16 bf16;
typedef __attribute__((ext_vector_type(16))) __bf16 bf16x16;
typedef __attribute__((ext_vector_type(8)))  float  floatx8;

__device__ __forceinline__ float sigf(float x) { return 1.0f / (1.0f + __expf(-x)); }
__device__ __forceinline__ float tanh_fast(float x) { return 2.0f * sigf(2.0f * x) - 1.0f; }

// ---------------- prep kernels ----------------
__global__ void f32_to_bf16_kernel(const float* __restrict__ in, bf16* __restrict__ out, int n) {
  int i = blockIdx.x * blockDim.x + threadIdx.x;
  if (i < n) out[i] = (bf16)in[i];
}

__global__ void bias_sum_kernel(const float* __restrict__ a, const float* __restrict__ b,
                                float* __restrict__ out, int n) {
  int i = blockIdx.x * blockDim.x + threadIdx.x;
  if (i < n) out[i] = a[i] + b[i];
}

// Pack W [N][K] (row-major f32) into bf16 A-fragment tiles:
// out[((nt*KT + kt)*32 + lane)*16 + e] = W[nt*16 + lane%16][kt*32 + kmap(lane,e)]
__global__ void pack_a_kernel(const float* __restrict__ W, bf16* __restrict__ out, int N, int K) {
  int o = blockIdx.x * blockDim.x + threadIdx.x;
  if (o >= N * K) return;
  int e    = o & 15;
  int lane = (o >> 4) & 31;
  int KT   = K >> 5;
  int kt   = (o >> 9) % KT;
  int nt   = o / (KT << 9);
  int n = nt * 16 + (lane & 15);
  int k = kt * 32 + ((lane >> 4) << 4) + e;
  out[o] = (bf16)W[(long)n * K + k];
}

// ---------------- GEMM building blocks ----------------
// Load 4 A-fragment tiles (tile idx tbase..tbase+3; idx = gate*2 + j) of k-tile kt.
__device__ __forceinline__ void loadA4(bf16x16* a, const bf16* __restrict__ pW,
                                       int KT, int wave, int lane, int kt, int tbase) {
#pragma unroll
  for (int i = 0; i < 4; ++i) {
    int idx = tbase + i;
    long nt = (long)((idx >> 1) * 32 + wave * 2 + (idx & 1));
    a[i] = *(const bf16x16*)(pW + ((nt * KT + kt) * 32 + lane) * 16);
  }
}

__device__ __forceinline__ void mma4(floatx8* acc, const bf16x16* a, bf16x16 b) {
#pragma unroll
  for (int i = 0; i < 4; ++i)
    acc[i] = __builtin_amdgcn_wmma_f32_16x16x32_bf16(
        false, a[i], false, b, (short)0, acc[i], false, false);
}

// Accumulate gates^T += W x in^T over K = KT*32, software-pipelined.
// USEG selects global (x sequence) vs LDS (h / y) B-operand at compile time
// so LDS stays in the ds_load path (no generic/flat pointers).
template <bool USEG>
__device__ __forceinline__ void gemm_gates(
    floatx8* acc, int lane, int wave,
    const bf16* __restrict__ pW, int KT,
    const bf16* __restrict__ inG, long inGStride,
    const unsigned char* __restrict__ inL, int inLStride)
{
  const int col16 = lane & 15;
  const int khalf = (lane >> 4) << 4;  // element offset 0 / 16
  bf16x16 aP[4], aQ[4], bCur, bNxt;

  if (USEG) bCur = *(const bf16x16*)(inG + (long)col16 * inGStride + khalf);
  else      bCur = *(const bf16x16*)(inL + col16 * inLStride + khalf * 2);
  loadA4(aP, pW, KT, wave, lane, 0, 0);

#pragma unroll 1
  for (int kt = 0; kt < KT; ++kt) {
    loadA4(aQ, pW, KT, wave, lane, kt, 4);            // 2nd half of current kt
    if (kt + 1 < KT) {                                 // B for next kt
      if (USEG) bNxt = *(const bf16x16*)(inG + (long)col16 * inGStride + (kt + 1) * 32 + khalf);
      else      bNxt = *(const bf16x16*)(inL + col16 * inLStride + ((kt + 1) * 32 + khalf) * 2);
    }
    mma4(acc + 0, aP, bCur);                           // overlaps aQ/bNxt loads
    if (kt + 1 < KT) loadA4(aP, pW, KT, wave, lane, kt + 1, 0);
    mma4(acc + 4, aQ, bCur);                           // overlaps aP loads
    bCur = bNxt;
  }
}

// One timestep of one LSTM layer. Wave owns h cols [wave*32, wave*32+32).
// D layout: lane = batch, VGPR r = gate row (r / r+8) -> lane-local update, c in regs.
template <bool USEG>
__device__ __forceinline__ void layer_step(
    int lane, int wave,
    const bf16* __restrict__ pWih, int KTin,
    const bf16* __restrict__ pWhh,
    const float* bsumL,                                 // LDS bias (bih+bhh)
    const bf16* __restrict__ inG, long inGStride,
    const unsigned char* __restrict__ inL, int inLStride,
    const unsigned char* __restrict__ hOld, unsigned char* __restrict__ hNew,
    float* c)
{
  floatx8 acc[8];
#pragma unroll
  for (int i = 0; i < 8; ++i)
#pragma unroll
    for (int q = 0; q < 8; ++q) acc[i][q] = 0.0f;

  gemm_gates<USEG>(acc, lane, wave, pWih, KTin, inG, inGStride, inL, inLStride);
  gemm_gates<false>(acc, lane, wave, pWhh, 16, (const bf16*)0, 0, hOld, HSTR);

  const int batch = lane & 15;
  const int rofs  = (lane >> 4) << 3;
#pragma unroll
  for (int j = 0; j < 2; ++j)
#pragma unroll
    for (int r = 0; r < 8; ++r) {
      int col  = wave * 32 + j * 16 + rofs + r;
      float gi = acc[0 + j][r] + bsumL[col];
      float gf = acc[2 + j][r] + bsumL[H_ + col];
      float gg = acc[4 + j][r] + bsumL[2 * H_ + col];
      float go = acc[6 + j][r] + bsumL[3 * H_ + col];
      float cn = sigf(gf) * c[j * 8 + r] + sigf(gi) * tanh_fast(gg);
      c[j * 8 + r] = cn;
      float h = sigf(go) * tanh_fast(cn);
      *(bf16*)(hNew + batch * HSTR + col * 2) = (bf16)h;
    }
}

// ---------------- main persistent kernel ----------------
__global__ __launch_bounds__(NTHR, 1) void lstm_ae_kernel(
    const bf16* __restrict__ xbf,
    const bf16* __restrict__ peWih0, const bf16* __restrict__ peWhh0,
    const bf16* __restrict__ peWih1, const bf16* __restrict__ peWhh1,
    const bf16* __restrict__ pdWih0, const bf16* __restrict__ pdWhh0,
    const bf16* __restrict__ pdWih1, const bf16* __restrict__ pdWhh1,
    const bf16* __restrict__ pWout,
    const float* __restrict__ bsum,        // [4][2048]
    const float* __restrict__ b_out,       // [64]
    const float* __restrict__ start_token, // [64]
    float* __restrict__ out)               // [B][T][I]
{
  __shared__ __align__(32) unsigned char h0buf[2][16 * HSTR];
  __shared__ __align__(32) unsigned char h1buf[2][16 * HSTR];
  __shared__ __align__(32) unsigned char ybuf[2][16 * YSTR];
  __shared__ float sbias[4 * G4_ + I_];   // 4 layer bias sums + b_out

  const int tid  = threadIdx.x;
  const int lane = tid & 31;
  const int wave = tid >> 5;
  const int b0   = blockIdx.x * 16;

  for (int i = tid * 4; i < 2 * 16 * HSTR; i += NTHR * 4) {
    *(int*)(&h0buf[0][0] + i) = 0;
    *(int*)(&h1buf[0][0] + i) = 0;
  }
  for (int i = tid * 4; i < 2 * 16 * YSTR; i += NTHR * 4) *(int*)(&ybuf[0][0] + i) = 0;
  for (int i = tid; i < 4 * G4_; i += NTHR) sbias[i] = bsum[i];
  if (tid < I_) sbias[4 * G4_ + tid] = b_out[tid];
  __syncthreads();
  if (tid < I_) {
    bf16 v = (bf16)start_token[tid];
    for (int b = 0; b < 16; ++b) *(bf16*)(&ybuf[0][0] + b * YSTR + tid * 2) = v;
  }
  float cA[16], cB[16];
#pragma unroll
  for (int i = 0; i < 16; ++i) { cA[i] = 0.0f; cB[i] = 0.0f; }
  __syncthreads();

  // ---- encoder: both layers pipelined per timestep ----
  for (int t = 0; t < T_; ++t) {
    int p = t & 1;
    layer_step<true>(lane, wave, peWih0, I_ / 32, peWhh0, sbias,
                     xbf + ((long)b0 * T_ + t) * I_, (long)T_ * I_, (const unsigned char*)0, 0,
                     h0buf[p], h0buf[p ^ 1], cA);
    __syncthreads();
    layer_step<false>(lane, wave, peWih1, H_ / 32, peWhh1, sbias + G4_,
                      (const bf16*)0, 0, h0buf[p ^ 1], HSTR,
                      h1buf[p], h1buf[p ^ 1], cB);
    __syncthreads();
  }
  // enc-final states: h0buf[0]/cA, h1buf[0]/cB -> decoder init (reference semantics).

  // ---- decoder: autoregressive with LDS feedback ----
  for (int t = 0; t < T_; ++t) {
    int p = t & 1;
    layer_step<false>(lane, wave, pdWih0, I_ / 32, pdWhh0, sbias + 2 * G4_,
                      (const bf16*)0, 0, ybuf[p], YSTR,
                      h0buf[p], h0buf[p ^ 1], cA);
    __syncthreads();
    layer_step<false>(lane, wave, pdWih1, H_ / 32, pdWhh1, sbias + 3 * G4_,
                      (const bf16*)0, 0, h0buf[p ^ 1], HSTR,
                      h1buf[p], h1buf[p ^ 1], cB);
    __syncthreads();
    // projection y = h1 @ W_out^T + b_out : 4 tiles, waves 0..3
    if (wave < 4) {
      floatx8 acc;
#pragma unroll
      for (int q = 0; q < 8; ++q) acc[q] = 0.0f;
      const int col16 = lane & 15;
      const int khalf = (lane >> 4) << 4;
      bf16x16 pa[2];
      pa[0] = *(const bf16x16*)(pWout + ((long)(wave * 16 + 0) * 32 + lane) * 16);
#pragma unroll 1
      for (int kt = 0; kt < 16; kt += 2) {
        pa[1] = *(const bf16x16*)(pWout + ((long)(wave * 16 + kt + 1) * 32 + lane) * 16);
        bf16x16 bf0 = *(const bf16x16*)(&h1buf[p ^ 1][0] + col16 * HSTR + (kt * 32 + khalf) * 2);
        acc = __builtin_amdgcn_wmma_f32_16x16x32_bf16(false, pa[0], false, bf0, (short)0, acc, false, false);
        if (kt + 2 < 16)
          pa[0] = *(const bf16x16*)(pWout + ((long)(wave * 16 + kt + 2) * 32 + lane) * 16);
        bf16x16 bf1 = *(const bf16x16*)(&h1buf[p ^ 1][0] + col16 * HSTR + ((kt + 1) * 32 + khalf) * 2);
        acc = __builtin_amdgcn_wmma_f32_16x16x32_bf16(false, pa[1], false, bf1, (short)0, acc, false, false);
      }
      const int batch = lane & 15;
      const int rofs  = (lane >> 4) << 3;
#pragma unroll
      for (int r = 0; r < 8; ++r) {
        int ocol = wave * 16 + rofs + r;
        float v = acc[r] + sbias[4 * G4_ + ocol];
        out[((long)(b0 + batch) * T_ + t) * I_ + ocol] = v;
        *(bf16*)(&ybuf[p ^ 1][0] + batch * YSTR + ocol * 2) = (bf16)v;
      }
    }
    __syncthreads();
  }
}

extern "C" void kernel_launch(void* const* d_in, const int* in_sizes, int n_in,
                              void* d_out, int out_size, void* d_ws, size_t ws_size,
                              hipStream_t stream) {
  (void)in_sizes; (void)n_in; (void)out_size; (void)ws_size;
  const float* x      = (const float*)d_in[0];
  const float* eWih0  = (const float*)d_in[1];
  const float* eWhh0  = (const float*)d_in[2];
  const float* ebih0  = (const float*)d_in[3];
  const float* ebhh0  = (const float*)d_in[4];
  const float* eWih1  = (const float*)d_in[5];
  const float* eWhh1  = (const float*)d_in[6];
  const float* ebih1  = (const float*)d_in[7];
  const float* ebhh1  = (const float*)d_in[8];
  const float* dWih0  = (const float*)d_in[9];
  const float* dWhh0  = (const float*)d_in[10];
  const float* dbih0  = (const float*)d_in[11];
  const float* dbhh0  = (const float*)d_in[12];
  const float* dWih1  = (const float*)d_in[13];
  const float* dWhh1  = (const float*)d_in[14];
  const float* dbih1  = (const float*)d_in[15];
  const float* dbhh1  = (const float*)d_in[16];
  const float* Wout   = (const float*)d_in[17];
  const float* b_out  = (const float*)d_in[18];
  const float* start  = (const float*)d_in[19];

  unsigned char* ws = (unsigned char*)d_ws;
  size_t off = 0;
  auto carve = [&](size_t bytes) -> void* {
    void* p = ws + off;
    off = (off + bytes + 255) & ~(size_t)255;
    return p;
  };
  bf16* xbf    = (bf16*)carve((size_t)B_ * T_ * I_ * 2);
  bf16* peWih0 = (bf16*)carve((size_t)G4_ * I_ * 2);
  bf16* peWhh0 = (bf16*)carve((size_t)G4_ * H_ * 2);
  bf16* peWih1 = (bf16*)carve((size_t)G4_ * H_ * 2);
  bf16* peWhh1 = (bf16*)carve((size_t)G4_ * H_ * 2);
  bf16* pdWih0 = (bf16*)carve((size_t)G4_ * I_ * 2);
  bf16* pdWhh0 = (bf16*)carve((size_t)G4_ * H_ * 2);
  bf16* pdWih1 = (bf16*)carve((size_t)G4_ * H_ * 2);
  bf16* pdWhh1 = (bf16*)carve((size_t)G4_ * H_ * 2);
  bf16* pWout  = (bf16*)carve((size_t)I_ * H_ * 2);
  float* bsum  = (float*)carve((size_t)4 * G4_ * 4);

  int n = B_ * T_ * I_;
  f32_to_bf16_kernel<<<(n + 255) / 256, 256, 0, stream>>>(x, xbf, n);

  pack_a_kernel<<<(G4_ * I_ + 255) / 256, 256, 0, stream>>>(eWih0, peWih0, G4_, I_);
  pack_a_kernel<<<(G4_ * H_ + 255) / 256, 256, 0, stream>>>(eWhh0, peWhh0, G4_, H_);
  pack_a_kernel<<<(G4_ * H_ + 255) / 256, 256, 0, stream>>>(eWih1, peWih1, G4_, H_);
  pack_a_kernel<<<(G4_ * H_ + 255) / 256, 256, 0, stream>>>(eWhh1, peWhh1, G4_, H_);
  pack_a_kernel<<<(G4_ * I_ + 255) / 256, 256, 0, stream>>>(dWih0, pdWih0, G4_, I_);
  pack_a_kernel<<<(G4_ * H_ + 255) / 256, 256, 0, stream>>>(dWhh0, pdWhh0, G4_, H_);
  pack_a_kernel<<<(G4_ * H_ + 255) / 256, 256, 0, stream>>>(dWih1, pdWih1, G4_, H_);
  pack_a_kernel<<<(G4_ * H_ + 255) / 256, 256, 0, stream>>>(dWhh1, pdWhh1, G4_, H_);
  pack_a_kernel<<<(I_ * H_ + 255) / 256, 256, 0, stream>>>(Wout, pWout, I_, H_);

  bias_sum_kernel<<<(G4_ + 255) / 256, 256, 0, stream>>>(ebih0, ebhh0, bsum + 0 * G4_, G4_);
  bias_sum_kernel<<<(G4_ + 255) / 256, 256, 0, stream>>>(ebih1, ebhh1, bsum + 1 * G4_, G4_);
  bias_sum_kernel<<<(G4_ + 255) / 256, 256, 0, stream>>>(dbih0, dbhh0, bsum + 2 * G4_, G4_);
  bias_sum_kernel<<<(G4_ + 255) / 256, 256, 0, stream>>>(dbih1, dbhh1, bsum + 3 * G4_, G4_);

  lstm_ae_kernel<<<B_ / 16, NTHR, 0, stream>>>(
      xbf, peWih0, peWhh0, peWih1, peWhh1,
      pdWih0, pdWhh0, pdWih1, pdWhh1, pWout,
      bsum, b_out, start, (float*)d_out);
}
